// Sequence2HybridMetric_89764816487070
// MI455X (gfx1250) — matrix operand order repaired and checked by reading, not compiled, not run on hardware
//
#include <hip/hip_runtime.h>
#include <hip/hip_bf16.h>
#include <math.h>

// ---------------- problem constants (match reference) ----------------
#define BB   32
#define SS   8192
#define DD   64            // embedding dim == GRU layer input dim (2U == 64 too)
#define UU   32
#define NG   96            // 3*U gate width
#define NV_  50000
#define EE   500000
#define MROWS (BB*SS)      // 262144

typedef __attribute__((ext_vector_type(16))) __bf16 v16bf;
typedef __attribute__((ext_vector_type(8)))  float  v8f;

// =====================================================================
// WMMA bf16 16x16x32 tile GEMM:  D[16,16] += A[16,Ktot] * B[Ktot,16]
// A: row-major f32, ld = ldx ; B: row-major f32 [Ktot, ldw] column slice
// Lane layouts per CDNA5 ISA §7.12.2 (wave32):
//   A 16-bit 16x32: lanes 0-15 -> M=lane, K in {0..7,16..23};
//                   lanes 16-31 -> M=lane-16, K in {8..15,24..31}
//   B 16-bit 32x16: lanes 0-15 -> N=lane, K in {0..7,16..23};
//                   lanes 16-31 -> N=lane-16, K in {8..15,24..31}
//   C/D f32 16x16:  VGPR v -> M = v + (lane<16 ? 0 : 8), N = lane&15
// =====================================================================
__device__ __forceinline__ v8f gemm_tile_bf16(const float* __restrict__ Xrow, int ldx,
                                              const float* __restrict__ Wcol, int ldw,
                                              int Ktot)
{
    const int lane = threadIdx.x & 31;
    const int m    = lane & 15;
    const int akb  = (lane < 16) ? 0 : 8;     // A K-subbase
    const int bkb  = (lane < 16) ? 0 : 8;     // B K-subbase
    v8f acc = {};
    for (int kc = 0; kc < Ktot; kc += 32) {
        v16bf a, b;
#pragma unroll
        for (int i = 0; i < 16; ++i) {
            const int v = i >> 1, h = i & 1;
            const int ka = (v < 4) ? (akb + 2*v + h) : (16 + akb + 2*(v-4) + h);
            a[i] = (__bf16)Xrow[(size_t)m * ldx + kc + ka];
            const int kb = (v < 4) ? (bkb + 2*v + h) : (16 + bkb + 2*(v-4) + h);
            b[i] = (__bf16)Wcol[(size_t)(kc + kb) * ldw + m];
        }
        acc = __builtin_amdgcn_wmma_f32_16x16x32_bf16(
                  /*neg_a=*/false, a, /*neg_b=*/false, b,
                  /*c_mod=*/(short)0, acc, /*reuse_a=*/false, /*reuse_b=*/false);
    }
    return acc;
}

__device__ __forceinline__ void store_tile(float* __restrict__ Orow, int ldo,
                                           const float* __restrict__ bias,
                                           int n0, v8f acc)
{
    const int lane  = threadIdx.x & 31;
    const int n     = lane & 15;
    const int mbase = (lane < 16) ? 0 : 8;
    const float bb  = bias ? bias[n0 + n] : 0.0f;
#pragma unroll
    for (int v = 0; v < 8; ++v)
        Orow[(size_t)(mbase + v) * ldo + n0 + n] = acc[v] + bb;
}

// ---------------------------------------------------------------------
// Input-projection GEMM: MX[dir][m, n] = X[m,:64] @ K[dir][:, n] + b_in[dir][n]
// grid: (Mtiles/4, 6, 2)  block: 128 (4 waves, one 16x16 tile each)
// ---------------------------------------------------------------------
__global__ void k_gemm_mx(const float* __restrict__ X,
                          const float* __restrict__ Wall,   // layer base of gru_kernels [2,64,96]
                          const float* __restrict__ Ball,   // layer base of gru_biases  [2,2,96]
                          float* __restrict__ MXall, int M)
{
    const int wave = threadIdx.x >> 5;
    const int mt   = blockIdx.x * 4 + wave;
    const int n0   = blockIdx.y * 16;
    const int dir  = blockIdx.z;
    if (mt * 16 >= M) return;
    const float* W   = Wall + (size_t)dir * DD * NG;
    const float* bin = Ball + (size_t)dir * 2 * NG;          // [0] = b_in
    float*       MX  = MXall + (size_t)dir * M * NG;
    v8f acc = gemm_tile_bf16(X + (size_t)mt * 16 * DD, DD, W + n0, NG, DD);
    store_tile(MX + (size_t)mt * 16 * NG, NG, bin, n0, acc);
}

// ---------------------------------------------------------------------
// Final linear: OUT[m,n] = VE[m,:64] @ Wlin[:, n]   (no bias)
// grid: (3125, 4)  block: 32
// ---------------------------------------------------------------------
__global__ void k_linear(const float* __restrict__ VE,
                         const float* __restrict__ Wlin,
                         float* __restrict__ OUT)
{
    const int mt = blockIdx.x;
    const int n0 = blockIdx.y * 16;
    v8f acc = gemm_tile_bf16(VE + (size_t)mt * 16 * DD, DD, Wlin + n0, DD, DD);
    store_tile(OUT + (size_t)mt * 16 * DD, DD, nullptr, n0, acc);
}

// ---------------------------------------------------------------------
// Sequential GRU recurrence. One workgroup per (batch, dir); 96 threads.
// Thread j owns gate column j: keeps R[:,j] (32 f32) in VGPRs.
// h[32] lives in LDS; mi[96] exchanged through LDS each step.
// ---------------------------------------------------------------------
__global__ void k_recur(const float* __restrict__ MXall,   // [2, B*S, 96]
                        const float* __restrict__ Rall,    // layer base of rec kernels [2,32,96]
                        const float* __restrict__ Ball,    // layer base of biases [2,2,96]
                        float* __restrict__ OUT)           // [B, S, 64]
{
    const int b   = blockIdx.x;
    const int dir = blockIdx.y;
    const int j   = threadIdx.x;                    // 0..95
    const float* R    = Rall + (size_t)dir * UU * NG;
    const float* brec = Ball + (size_t)dir * 2 * NG + NG;   // [1] = b_rec
    const float* MX   = MXall + (size_t)dir * MROWS * NG + (size_t)b * SS * NG;

    float Rcol[UU];
#pragma unroll
    for (int k = 0; k < UU; ++k) Rcol[k] = R[k * NG + j];
    const float bj = brec[j];

    __shared__ float h[UU];
    __shared__ float mi[NG];
    if (j < UU) h[j] = 0.0f;
    __syncthreads();

    for (int t = 0; t < SS; ++t) {
        const int s = dir ? (SS - 1 - t) : t;
        float acc = bj;
#pragma unroll
        for (int k = 0; k < UU; ++k) acc += h[k] * Rcol[k];
        mi[j] = acc;
        __syncthreads();
        if (j < UU) {
            const float* mx = MX + (size_t)s * NG;
            const float z  = 1.0f / (1.0f + __expf(-(mx[j]        + mi[j])));
            const float r  = 1.0f / (1.0f + __expf(-(mx[UU + j]   + mi[UU + j])));
            const float hh = tanhf(mx[2*UU + j] + r * mi[2*UU + j]);
            const float hn = z * h[j] + (1.0f - z) * hh;
            OUT[((size_t)b * SS + s) * (2*UU) + dir * UU + j] = hn;
            h[j] = hn;
        }
        __syncthreads();
    }
}

// ---------------------------------------------------------------------
// Elementwise / scatter-gather helpers
// ---------------------------------------------------------------------
__global__ void k_embed(const float* __restrict__ table, const int* __restrict__ gm,
                        const int* __restrict__ slen, float* __restrict__ out)
{
    const size_t i = (size_t)blockIdx.x * blockDim.x + threadIdx.x;
    if (i >= (size_t)MROWS * DD) return;
    const int    dd = (int)(i & (DD - 1));
    const size_t bs = i >> 6;
    const int    s  = (int)(bs & (SS - 1));
    const int    b  = (int)(bs >> 13);
    const int    tok = gm[bs];
    const float  v = table[(size_t)tok * DD + dd];
    out[i] = (s < slen[b]) ? v : 0.0f;
}

__global__ void k_mask(float* __restrict__ cur, const int* __restrict__ slen)
{
    const size_t i = (size_t)blockIdx.x * blockDim.x + threadIdx.x;
    if (i >= (size_t)MROWS * DD) return;
    const size_t bs = i >> 6;
    const int    s  = (int)(bs & (SS - 1));
    const int    b  = (int)(bs >> 13);
    if (s >= slen[b]) cur[i] = 0.0f;
}

__global__ void k_zero(float* __restrict__ p, size_t n)
{
    const size_t i = (size_t)blockIdx.x * blockDim.x + threadIdx.x;
    if (i < n) p[i] = 0.0f;
}

__global__ void k_segsum(const float* __restrict__ cur, const int* __restrict__ tok,
                         const int* __restrict__ var, float* __restrict__ sums,
                         float* __restrict__ cnt)
{
    const size_t i = (size_t)blockIdx.x * blockDim.x + threadIdx.x;
    if (i >= (size_t)EE * 16) return;
    const int    c = (int)(i & 15);
    const size_t e = i >> 4;
    const int t = tok[e], v = var[e];
    const float* src = cur  + (size_t)t * DD + c * 4;
    float*       dst = sums + (size_t)v * DD + c * 4;
#pragma unroll
    for (int q = 0; q < 4; ++q) atomicAdd(dst + q, src[q]);
    if (c == 0) atomicAdd(cnt + v, 1.0f);
}

__global__ void k_mean(const float* __restrict__ sums, const float* __restrict__ cnt,
                       float* __restrict__ out)
{
    const size_t i = (size_t)blockIdx.x * blockDim.x + threadIdx.x;
    if (i >= (size_t)NV_ * DD) return;
    out[i] = sums[i] / fmaxf(cnt[i >> 6], 1.0f);
}

__global__ void k_scatter(float* __restrict__ cur, const int* __restrict__ tok,
                          const int* __restrict__ var, const float* __restrict__ ve)
{
    const size_t i = (size_t)blockIdx.x * blockDim.x + threadIdx.x;
    if (i >= (size_t)EE * 16) return;
    const int    c = (int)(i & 15);
    const size_t e = i >> 4;
    const int t = tok[e], v = var[e];
    float*       dst = cur + (size_t)t * DD + c * 4;
    const float* src = ve  + (size_t)v * DD + c * 4;
#pragma unroll
    for (int q = 0; q < 4; ++q) atomicAdd(dst + q, src[q]);
}

// =====================================================================
extern "C" void kernel_launch(void* const* d_in, const int* in_sizes, int n_in,
                              void* d_out, int out_size, void* d_ws, size_t ws_size,
                              hipStream_t stream)
{
    (void)in_sizes; (void)n_in; (void)out_size; (void)ws_size;
    const float* emb  = (const float*)d_in[0];   // [10000,64]
    const float* gk   = (const float*)d_in[1];   // [2,2,64,96]
    const float* grk  = (const float*)d_in[2];   // [2,2,32,96]
    const float* gb   = (const float*)d_in[3];   // [2,2,2,96]
    const float* lw   = (const float*)d_in[4];   // [64,64]
    const int*   gmat = (const int*)d_in[5];     // [32,8192]
    const int*   slen = (const int*)d_in[6];     // [32]
    const int*   vbt  = (const int*)d_in[7];     // [E]
    const int*   tvi  = (const int*)d_in[8];     // [E]
    float* out = (float*)d_out;                  // [NV,64] ++ [NV,64]

    // workspace layout (floats)
    float* curA = (float*)d_ws;                         // 16777216
    float* curB = curA + (size_t)MROWS * DD;            // 16777216
    float* mx   = curB + (size_t)MROWS * DD;            // 2 * 262144 * 96 = 50331648
    float* sums = mx   + (size_t)2 * MROWS * NG;        // 3200000
    float* cnt  = sums + (size_t)NV_ * DD;              // 50000

    const int elemN  = MROWS * DD;                      // 16777216
    const int elemB  = (elemN + 255) / 256;             // 65536
    const int segB   = (EE * 16 + 255) / 256;           // 31250
    const int meanB  = (NV_ * DD + 255) / 256;          // 12500
    const int zeroN  = NV_ * DD + NV_;
    const int zeroB  = (zeroN + 255) / 256;
    const dim3 gemmG(MROWS / 16 / 4, NG / 16, 2);       // (4096, 6, 2)

    // 1) embedding gather + mask -> curA
    k_embed<<<elemB, 256, 0, stream>>>(emb, gmat, slen, curA);

    // ------------- layer 0 -------------
    k_gemm_mx<<<gemmG, 128, 0, stream>>>(curA, gk, gb, mx, MROWS);
    k_recur<<<dim3(BB, 2), NG, 0, stream>>>(mx, grk, gb, curB);

    // consistency layer: segment mean + scatter back into curB
    k_zero<<<zeroB, 256, 0, stream>>>(sums, (size_t)zeroN);
    k_segsum<<<segB, 256, 0, stream>>>(curB, vbt, tvi, sums, cnt);
    k_mean<<<meanB, 256, 0, stream>>>(sums, cnt, sums);           // in-place -> ve
    k_scatter<<<segB, 256, 0, stream>>>(curB, vbt, tvi, sums);
    k_mask<<<elemB, 256, 0, stream>>>(curB, slen);

    // ------------- layer 1 -------------
    k_gemm_mx<<<gemmG, 128, 0, stream>>>(curB, gk + 2*DD*NG, gb + 2*2*NG, mx, MROWS);
    k_recur<<<dim3(BB, 2), NG, 0, stream>>>(mx, grk + 2*UU*NG, gb + 2*2*NG, curA);

    // final variable embeddings -> out[0 : NV*64]
    k_zero<<<zeroB, 256, 0, stream>>>(sums, (size_t)zeroN);
    k_segsum<<<segB, 256, 0, stream>>>(curA, vbt, tvi, sums, cnt);
    k_mean<<<meanB, 256, 0, stream>>>(sums, cnt, out);

    // classification = ve @ linear_w -> out[NV*64 : 2*NV*64]
    k_linear<<<dim3(NV_ / 16, DD / 16), 32, 0, stream>>>(out, lw, out + (size_t)NV_ * DD);
}
